// LSTMClassifier_7799660609981
// MI455X (gfx1250) — matrix operand order repaired and checked by reading, not compiled, not run on hardware
//
#include <hip/hip_runtime.h>
#include <hip/hip_bf16.h>

// ---------------------------------------------------------------------------
// LSTMClassifier on gfx1250 (MI455X)
//   VOCAB=50257, EMB=48, HID=128, 4H=512, NCLS=2000, B=512, S=1024
// Strategy:
//   P0: convert w_ih (pad K 48->64) and w_hh to f16
//   P1: table[v,:] = emb[v] @ w_ih^T + b_ih + b_hh   (WMMA, f16 table, 51MB -> L2)
//   P2: persistent LSTM scan, 32 WGs x 512 thr, WMMA f16, w_hh in registers,
//       software-pipelined xg gather
//   P3: FC  out = h_n @ w_fc^T + b_fc
// ---------------------------------------------------------------------------

#define VOCAB 50257
#define EMB   48
#define HID   128
#define G4    512          // 4*HID
#define NCLS  2000
#define BATCH 512
#define SEQ   1024

#define GSTRIDE 516        // f32 stride for g[16][512] in LDS (bank-conflict pad)
#define HSTRIDE 136        // f16 stride for h[16][128] in LDS (16B-aligned pad)

typedef __attribute__((ext_vector_type(16))) _Float16 v16h;
typedef __attribute__((ext_vector_type(8)))  _Float16 v8h;
typedef __attribute__((ext_vector_type(4)))  _Float16 v4h;
typedef __attribute__((ext_vector_type(8)))  float    v8f;

__device__ __forceinline__ v16h cat8(v8h lo, v8h hi) {
    return __builtin_shufflevector(lo, hi, 0,1,2,3,4,5,6,7,8,9,10,11,12,13,14,15);
}

__device__ __forceinline__ float fast_tanh(float x) {
#if __has_builtin(__builtin_amdgcn_tanhf)
    return __builtin_amdgcn_tanhf(x);          // v_tanh_f32 (gfx1250)
#else
    float e = __expf(2.0f * x);
    return (e - 1.0f) / (e + 1.0f);
#endif
}
__device__ __forceinline__ float fast_sigmoid(float x) {
    return 0.5f + 0.5f * fast_tanh(0.5f * x);  // 1 trans op per sigmoid
}

// ---------------------------------------------------------------------------
// P0: weight conversion.  wih16: [512][64] f16 (cols 48..63 = 0).  whh16: [512][128] f16.
// ---------------------------------------------------------------------------
__global__ void prep_weights(const float* __restrict__ w_ih, const float* __restrict__ w_hh,
                             _Float16* __restrict__ wih16, _Float16* __restrict__ whh16) {
    int idx = blockIdx.x * 256 + threadIdx.x;
    const int NIH = G4 * 64;            // 32768 padded w_ih elems
    if (idx < NIH) {
        int row = idx >> 6, col = idx & 63;
        wih16[idx] = (col < EMB) ? (_Float16)w_ih[row * EMB + col] : (_Float16)0.0f;
    } else {
        int j = idx - NIH;              // w_hh elems: 512*128 = 65536
        if (j < G4 * HID) whh16[j] = (_Float16)w_hh[j];
    }
}

// ---------------------------------------------------------------------------
// P1: per-token gate-preactivation table: table[v][n] = bias[n] + sum_e emb[v][e]*w_ih[n][e]
//     Grid: ceil(VOCAB/16) blocks x 256 threads (8 waves). Wave owns 4 N-tiles. K=64 (2 frags).
// ---------------------------------------------------------------------------
__global__ __launch_bounds__(256, 1)
void build_table(const float* __restrict__ emb, const _Float16* __restrict__ wih16,
                 const float* __restrict__ b_ih, const float* __restrict__ b_hh,
                 _Float16* __restrict__ table) {
    const int lane = threadIdx.x & 31;
    const int wave = threadIdx.x >> 5;        // 0..7
    const int lm = lane & 15, lh = lane >> 4; // lh: half-wave select
    const int v0 = blockIdx.x * 16;
    const bool full = (v0 + 16) <= VOCAB;     // block-uniform: all rows valid?

    // A fragments: emb rows (f32 -> f16), shared across this wave's tiles
    int vrow = v0 + lm; if (vrow > VOCAB - 1) vrow = VOCAB - 1;
    const float* er = emb + (size_t)vrow * EMB;
    v16h A[2];
#pragma unroll
    for (int kf = 0; kf < 2; ++kf) {
        int clo = 32 * kf + 8 * lh;           // vgpr0-3: K = clo..clo+7
        int chi = clo + 16;                   // vgpr4-7: K = chi..chi+7
#pragma unroll
        for (int i = 0; i < 8; ++i) {
            A[kf][i]     = (clo + i < EMB) ? (_Float16)er[clo + i] : (_Float16)0.0f;
            A[kf][8 + i] = (chi + i < EMB) ? (_Float16)er[chi + i] : (_Float16)0.0f;
        }
    }

#pragma unroll
    for (int tt = 0; tt < 4; ++tt) {
        const int n0 = wave * 64 + tt * 16;
        const int n  = n0 + lm;
        // B fragments from padded wih16 [512][64]
        v16h Bf[2];
#pragma unroll
        for (int kf = 0; kf < 2; ++kf) {
            const v8h* p = (const v8h*)(wih16 + n * 64 + 32 * kf + 16 * lh);
            Bf[kf] = cat8(*(const v8h*)(wih16 + n * 64 + 32 * kf + 8 * lh), *(const v8h*)(wih16 + n * 64 + 32 * kf + 16 + 8 * lh));
        }
        float bv = b_ih[n] + b_hh[n];
        v8f acc;
#pragma unroll
        for (int i = 0; i < 8; ++i) acc[i] = bv;
        acc = __builtin_amdgcn_wmma_f32_16x16x32_f16(false, A[0], false, Bf[0], (short)0, acc, false, false);
        acc = __builtin_amdgcn_wmma_f32_16x16x32_f16(false, A[1], false, Bf[1], (short)0, acc, false, false);

        _Float16* tcol = table + (size_t)(v0 + 8 * lh) * G4 + n;
        if (full) {                           // branch-free fast path (all but last block)
#pragma unroll
            for (int r = 0; r < 8; ++r)
                tcol[(size_t)r * G4] = (_Float16)acc[r];
        } else {
#pragma unroll
            for (int r = 0; r < 8; ++r)
                if (v0 + r + 8 * lh < VOCAB) tcol[(size_t)r * G4] = (_Float16)acc[r];
        }
    }
}

// ---------------------------------------------------------------------------
// P2: persistent LSTM scan. Grid: 32 blocks (batch tiles of 16) x 512 threads (16 waves).
//     Wave owns 2 N-tiles of the 16x512 gate matrix; w_hh f16 fragments live in VGPRs.
//     xg gather for step t+1 is issued right after step t's WMMAs (latency hides
//     behind the barrier + transcendental gate phase).
// ---------------------------------------------------------------------------
__device__ __forceinline__ void gather_xg(v8f nxt[2], const int* __restrict__ x,
                                          const _Float16* __restrict__ table,
                                          int b0, int t, int wave, int lm, int lh) {
    int tok = x[(size_t)(b0 + lm) * SEQ + t];   // lanes 0..15 hold row lm; 16..31 dup
#pragma unroll
    for (int r = 0; r < 8; ++r) {
        int tr = __shfl(tok, r + 8 * lh, 32);   // token of row M = r + 8*lh
        const _Float16* trow = table + (size_t)tr * G4;
#pragma unroll
        for (int tt = 0; tt < 2; ++tt)
            nxt[tt][r] = (float)trow[wave * 32 + tt * 16 + lm];
    }
}

__global__ __launch_bounds__(512, 1)
void lstm_scan(const int* __restrict__ x, const _Float16* __restrict__ table,
               const _Float16* __restrict__ whh16, float* __restrict__ hn) {
    __shared__ float    g[16 * GSTRIDE];      // 33 KB: gate preactivations [16][512]
    __shared__ _Float16 hb[16 * HSTRIDE];     // 4.25 KB: h as f16 A-matrix source

    const int tid  = threadIdx.x;
    const int lane = tid & 31;
    const int wave = tid >> 5;                // 0..15
    const int lm = lane & 15, lh = lane >> 4;
    const int b0 = blockIdx.x * 16;

    // --- B fragments (w_hh^T) resident in registers for the whole scan ---
    v16h Bf[2][4];                            // [n-tile][k-frag], 64 VGPRs
#pragma unroll
    for (int tt = 0; tt < 2; ++tt) {
        int row = wave * 32 + tt * 16 + lm;   // gate unit index n (B column)
#pragma unroll
        for (int kf = 0; kf < 4; ++kf) {
            const v8h* p = (const v8h*)(whh16 + row * HID + 32 * kf + 16 * lh);
            Bf[tt][kf] = cat8(*(const v8h*)(whh16 + row * HID + 32 * kf + 8 * lh), *(const v8h*)(whh16 + row * HID + 32 * kf + 16 + 8 * lh));
        }
    }

    // --- init: h = 0 in LDS, c = 0 in registers (4 cells per thread) ---
    for (int i = tid; i < 16 * HSTRIDE; i += 512) hb[i] = (_Float16)0.0f;
    const int gm = tid >> 5;                  // batch row owned in gate phase (0..15)
    const int gn = (tid & 31) * 4;            // 4 consecutive hidden units
    float c0[4];
#pragma unroll
    for (int i = 0; i < 4; ++i) c0[i] = 0.0f;

    // pipeline prologue: gather xg for t=0
    v8f nxt[2];
    gather_xg(nxt, x, table, b0, 0, wave, lm, lh);
    __syncthreads();

    for (int t = 0; t < SEQ; ++t) {
        // accumulators start from the prefetched xg
        v8f acc[2];
#pragma unroll
        for (int tt = 0; tt < 2; ++tt) acc[tt] = nxt[tt];

        // A fragments: h[16][128] from LDS (2 x ds_load_b128 per fragment)
        v16h A[4];
        const _Float16* hrow = hb + lm * HSTRIDE;
#pragma unroll
        for (int kf = 0; kf < 4; ++kf) {
            v8h lo = *(const v8h*)(hrow + 32 * kf + 8 * lh);
            v8h hi = *(const v8h*)(hrow + 32 * kf + 16 + 8 * lh);
            A[kf] = cat8(lo, hi);
        }

        // g_tile = xg_tile + h @ w_hh^T_tile  (K=128; kf outer so A is reused
        // by consecutive WMMAs; per-tile accumulation order unchanged)
#pragma unroll
        for (int kf = 0; kf < 4; ++kf)
#pragma unroll
            for (int tt = 0; tt < 2; ++tt)
                acc[tt] = __builtin_amdgcn_wmma_f32_16x16x32_f16(
                    false, A[kf], false, Bf[tt][kf], (short)0, acc[tt], false, false);

        // prefetch next step's xg (overlaps g-spill, barrier, gate phase)
        int tp = (t + 1 < SEQ) ? t + 1 : SEQ - 1;
        gather_xg(nxt, x, table, b0, tp, wave, lm, lh);

        // spill gate preactivations to LDS
#pragma unroll
        for (int tt = 0; tt < 2; ++tt) {
            int n = wave * 32 + tt * 16 + lm;
#pragma unroll
            for (int r = 0; r < 8; ++r)
                g[(r + 8 * lh) * GSTRIDE + n] = acc[tt][r];
        }
        __syncthreads();

        // gate phase: c stays in registers, h -> LDS f16 for next step's A
        {
            const float* grow = g + gm * GSTRIDE + gn;
            _Float16 hloc[4];
#pragma unroll
            for (int i = 0; i < 4; ++i) {
                float gi = grow[i];
                float gf = grow[128 + i];
                float gg = grow[256 + i];
                float go = grow[384 + i];
                float si = fast_sigmoid(gi);
                float sf = fast_sigmoid(gf);
                float so = fast_sigmoid(go);
                float tg = fast_tanh(gg);
                float cn = sf * c0[i] + si * tg;
                c0[i] = cn;
                float h = so * fast_tanh(cn);
                hloc[i] = (_Float16)h;
                if (t == SEQ - 1) hn[(size_t)(b0 + gm) * HID + gn + i] = h;
            }
            *(v4h*)(hb + gm * HSTRIDE + gn) = *(v4h*)hloc;
        }
        __syncthreads();
    }
}

// ---------------------------------------------------------------------------
// P3: out[b][j] = b_fc[j] + dot(hn[b], w_fc[j])   (0.26 GFLOP, trivial)
// ---------------------------------------------------------------------------
__global__ void fc_kernel(const float* __restrict__ hn, const float* __restrict__ w_fc,
                          const float* __restrict__ b_fc, float* __restrict__ out) {
    int idx = blockIdx.x * 256 + threadIdx.x;
    if (idx >= BATCH * NCLS) return;
    int b = idx / NCLS, j = idx - b * NCLS;
    const float4* hv = (const float4*)(hn + (size_t)b * HID);
    const float4* wv = (const float4*)(w_fc + (size_t)j * HID);
    float s = b_fc[j];
#pragma unroll
    for (int k = 0; k < HID / 4; ++k) {
        float4 h4 = hv[k], w4 = wv[k];
        s += h4.x * w4.x + h4.y * w4.y + h4.z * w4.z + h4.w * w4.w;
    }
    out[idx] = s;
}

// ---------------------------------------------------------------------------
extern "C" void kernel_launch(void* const* d_in, const int* in_sizes, int n_in,
                              void* d_out, int out_size, void* d_ws, size_t ws_size,
                              hipStream_t stream) {
    const int*   x     = (const int*)  d_in[0];
    const float* emb   = (const float*)d_in[1];
    const float* w_ih  = (const float*)d_in[2];
    const float* w_hh  = (const float*)d_in[3];
    const float* b_ih  = (const float*)d_in[4];
    const float* b_hh  = (const float*)d_in[5];
    const float* w_fc  = (const float*)d_in[6];
    const float* b_fc  = (const float*)d_in[7];
    float* out = (float*)d_out;

    // workspace carve-up (~52 MB total)
    char* ws = (char*)d_ws;
    size_t off = 0;
    _Float16* table = (_Float16*)(ws + off); off += (size_t)VOCAB * G4 * sizeof(_Float16);
    off = (off + 255) & ~(size_t)255;
    _Float16* wih16 = (_Float16*)(ws + off); off += (size_t)G4 * 64  * sizeof(_Float16);
    _Float16* whh16 = (_Float16*)(ws + off); off += (size_t)G4 * HID * sizeof(_Float16);
    off = (off + 255) & ~(size_t)255;
    float*    hn    = (float*)   (ws + off); off += (size_t)BATCH * HID * sizeof(float);
    (void)ws_size; (void)in_sizes; (void)n_in; (void)out_size;

    // P0: weight conversion
    {
        int total = G4 * 64 + G4 * HID;
        prep_weights<<<(total + 255) / 256, 256, 0, stream>>>(w_ih, w_hh, wih16, whh16);
    }
    // P1: token -> gate-preactivation table (WMMA)
    {
        int vt = (VOCAB + 15) / 16;
        build_table<<<vt, 256, 0, stream>>>(emb, wih16, b_ih, b_hh, table);
    }
    // P2: persistent recurrent scan (WMMA)
    lstm_scan<<<BATCH / 16, 512, 0, stream>>>(x, table, whh16, hn);
    // P3: classifier head
    fc_kernel<<<(BATCH * NCLS + 255) / 256, 256, 0, stream>>>(hn, w_fc, b_fc, out);
}
